// MyConv2dPac_5463198401055
// MI455X (gfx1250) — matrix operand order repaired and checked by reading, not compiled, not run on hardware
//
#include <hip/hip_runtime.h>

typedef __attribute__((ext_vector_type(16))) __bf16          v16bf;
typedef __attribute__((ext_vector_type(8)))  float           v8f;
typedef __attribute__((ext_vector_type(16))) unsigned short  v16u;
typedef __attribute__((ext_vector_type(8)))  unsigned int    v8ui;

#define BATCH 8
#define CIN   64
#define HH    128
#define WW    128
#define OUTC  64
#define NT    64        // pixels (N) per block: one row segment
#define NTAP  9
#define CSTR  80        // padded channel stride in x tile (halves): 32B aligned, bank-spread
#define XTCOLS 66       // 64 + 2 halo columns
// packed-weight scratch: [tap][mblk][ks][lane][16 halves] in WMMA A-fragment order
#define WPK_ELEMS (NTAP * 4 * 2 * 32 * 16)   // 36864 halves = 73728 bytes in d_ws

__device__ __forceinline__ unsigned short f2bf(float f) {
    unsigned int u = __float_as_uint(f);
    u += 0x7FFFu + ((u >> 16) & 1u);          // round-to-nearest-even
    return (unsigned short)(u >> 16);
}

// ---------------- kernel 1: pack weights into WMMA A-fragment lane order ----
__global__ __launch_bounds__(256)
void pac_pack_weights(const float* __restrict__ weight,
                      unsigned short* __restrict__ wpk) {
  int idx = blockIdx.x * 256 + threadIdx.x;
  if (idx >= WPK_ELEMS) return;
  int hh   = idx & 15;
  int lane = (idx >> 4) & 31;
  int ks   = (idx >> 9) & 1;
  int mblk = (idx >> 11) & 3;
  int t    = idx >> 13;
  // A k order used here (B frags match): lanes0-15 K = k0..k0+15 (k0=ks*32)
  //                               lanes16-31 same rows, K shifted by 16
  int m = mblk * 16 + (lane & 15);
  int c = ks * 32 + ((lane >> 4) << 4) + hh;
  wpk[idx] = f2bf(weight[(m * CIN + c) * NTAP + t]);
}

// ---------------- kernel 2: PAC as implicit GEMM on bf16 WMMA -----------------
__global__ __launch_bounds__(256)
void pac_wmma_kernel(const float* __restrict__ x,
                     const float* __restrict__ guide,
                     const unsigned short* __restrict__ wpk,
                     const float* __restrict__ bias,
                     float* __restrict__ out) {
  // x halo tile, channel-contiguous: [3 rows][66 cols][64 ch padded to 80] bf16
  __shared__ __align__(32) unsigned short xTs[3 * XTCOLS * CSTR];  // 31680 B
  __shared__ unsigned int gaussP[NT * 12];   // packed bf16 pair (g<<16)|g per (pixel, tap)
  __shared__ float biasS[OUTC];

  const int tid  = threadIdx.x;
  const int lane = tid & 31;
  const int wid  = tid >> 5;
  const int b    = blockIdx.z;
  const int h    = blockIdx.y;
  const int w0   = blockIdx.x * NT;

  if (tid < OUTC) biasS[tid] = bias[tid];

  // per-pixel 9-tap Gaussian from the 3-channel guide, stored as packed bf16 pair
  if (tid < NT) {
    const int w = w0 + tid;
    float c0 = guide[((b * 3 + 0) * HH + h) * WW + w];
    float c1 = guide[((b * 3 + 1) * HH + h) * WW + w];
    float c2 = guide[((b * 3 + 2) * HH + h) * WW + w];
    #pragma unroll
    for (int t = 0; t < NTAP; ++t) {
      int hh = h + t / 3 - 1, ww = w + t % 3 - 1;
      float g = 0.f;
      if (hh >= 0 && hh < HH && ww >= 0 && ww < WW) {
        float d0 = guide[((b * 3 + 0) * HH + hh) * WW + ww] - c0;
        float d1 = guide[((b * 3 + 1) * HH + hh) * WW + ww] - c1;
        float d2 = guide[((b * 3 + 2) * HH + hh) * WW + ww] - c2;
        g = __expf(-0.5f * (d0 * d0 + d1 * d1 + d2 * d2));
      }
      unsigned int gu = (unsigned int)f2bf(g);
      gaussP[tid * 12 + t] = (gu << 16) | gu;
    }
  }

  // stage x halo tile once (zero-padded), channel axis contiguous for B frags
  for (int e = tid; e < CIN * 3 * XTCOLS; e += 256) {
    int col = e % XTCOLS;
    int r   = (e / XTCOLS) % 3;
    int c   = e / (3 * XTCOLS);
    int hh = h + r - 1, ww = w0 + col - 1;
    float v = 0.f;
    if (hh >= 0 && hh < HH && ww >= 0 && ww < WW)
      v = x[((b * CIN + c) * HH + hh) * WW + ww];
    xTs[(r * XTCOLS + col) * CSTR + c] = f2bf(v);
  }
  __syncthreads();

  // wave -> 2 subtiles: m-block (16 rows of O), two n-blocks of 16 pixels
  const int mblk  = wid & 3;
  const int mBase = mblk * 16;
  const int nb    = (wid >> 2) * 32;
  const int nlo   = lane & 15;
  const int khalf = lane >> 4;            // 0: lanes 0-15, 1: lanes 16-31
  v8f acc0 = {};
  v8f acc1 = {};

  const v16u* __restrict__ wfrag = (const v16u*)wpk;

  #pragma unroll
  for (int t = 0; t < NTAP; ++t) {
    const int dh = t / 3, dw = t % 3;
    // per-pixel Gaussian, already packed as a bf16 pair: one ds_load_b32 each
    const unsigned int p0 = gaussP[(nb + nlo) * 12 + t];
    const unsigned int p1 = gaussP[(nb + nlo + 16) * 12 + t];
    const v8ui gv0i = {p0, p0, p0, p0, p0, p0, p0, p0};
    const v8ui gv1i = {p1, p1, p1, p1, p1, p1, p1, p1};
    const v16bf gv0 = __builtin_bit_cast(v16bf, gv0i);
    const v16bf gv1 = __builtin_bit_cast(v16bf, gv1i);

    #pragma unroll
    for (int ks = 0; ks < 2; ++ks) {
      // A fragment: one 32B load from packed, L2-resident weights
      int fidx = ((t * 4 + mblk) * 2 + ks) * 32 + lane;
      v16bf a = __builtin_bit_cast(v16bf, wfrag[fidx]);
      if (t < NTAP - 1)
        __builtin_prefetch(&wfrag[fidx + 8 * 32], 0, 3);

      // B fragments: lane = pixel column, 16 contiguous channels from x tile
      int kstart = ks * 32 + (khalf << 4);
      int n0 = nb + nlo;
      v16bf xb0 = __builtin_bit_cast(v16bf,
          *(const v16u*)&xTs[(dh * XTCOLS + n0 + dw) * CSTR + kstart]);
      v16bf xb1 = __builtin_bit_cast(v16bf,
          *(const v16u*)&xTs[(dh * XTCOLS + n0 + 16 + dw) * CSTR + kstart]);

      // per-pixel Gaussian modulation: v_pk_mul_bf16 splat multiply
      v16bf b0 = xb0 * gv0;
      v16bf b1 = xb1 * gv1;

      acc0 = __builtin_amdgcn_wmma_f32_16x16x32_bf16(false, a, false, b0,
                                                     (short)0, acc0, false, false);
      acc1 = __builtin_amdgcn_wmma_f32_16x16x32_bf16(false, a, false, b1,
                                                     (short)0, acc1, false, false);
    }
  }

  // epilogue: D layout -> lanes 0-15: M=r, N=lane; lanes 16-31: M=8+r, N=lane-16
  const int mlo = mBase + (khalf << 3);
  #pragma unroll
  for (int r = 0; r < 8; ++r) {
    int o = mlo + r;
    float bv = biasS[o];
    out[((b * OUTC + o) * HH + h) * WW + (w0 + nb + nlo)]      = acc0[r] + bv;
    out[((b * OUTC + o) * HH + h) * WW + (w0 + nb + 16 + nlo)] = acc1[r] + bv;
  }
}

extern "C" void kernel_launch(void* const* d_in, const int* in_sizes, int n_in,
                              void* d_out, int out_size, void* d_ws, size_t ws_size,
                              hipStream_t stream) {
  (void)in_sizes; (void)n_in; (void)out_size; (void)ws_size;  // needs 73728 B of d_ws
  const float* x      = (const float*)d_in[0];
  const float* guide  = (const float*)d_in[1];
  const float* weight = (const float*)d_in[2];
  const float* bias   = (const float*)d_in[3];
  unsigned short* wpk = (unsigned short*)d_ws;
  float* out          = (float*)d_out;

  // 1) pack weights into WMMA A-fragment order (stream-ordered, deterministic)
  pac_pack_weights<<<(WPK_ELEMS + 255) / 256, 256, 0, stream>>>(weight, wpk);
  // 2) main PAC implicit-GEMM kernel
  dim3 grid(WW / NT, HH, BATCH);   // (2, 128, 8)
  pac_wmma_kernel<<<grid, dim3(256), 0, stream>>>(x, guide, wpk, bias, out);
}
